// Decoder_2714419331668
// MI455X (gfx1250) — compile-verified, matrix operand order checked by reading, not run on hardware
//
#include <hip/hip_runtime.h>
#include <hip/hip_bf16.h>
#include <math.h>

typedef _Float16 f16;
typedef __attribute__((ext_vector_type(16))) _Float16 v16h;
typedef __attribute__((ext_vector_type(8)))  _Float16 v8h;
typedef __attribute__((ext_vector_type(8)))  float    v8f;

#define B_   64
#define S_   512
#define D_   64
#define H_   4
#define HD_  16
#define A_   32
#define M_   64
#define KTOP 8
#define HID_ 256
#define NB_  4
#define BS_  (B_*S_)
#define QKVS 192
#define EPS_ 1e-5f

// ---------------------------------------------------------------------------
// DPP-based wave reductions.
// quad_perm(1,0,3,2)=0xB1, quad_perm(2,3,0,1)=0x4E, row_ror:4=0x124, row_ror:8=0x128
// ---------------------------------------------------------------------------
template<int CTRL>
__device__ __forceinline__ float dppf(float v) {
    return __builtin_bit_cast(float, __builtin_amdgcn_update_dpp(
        0, __builtin_bit_cast(int, v), CTRL, 0xF, 0xF, true));
}
template<int CTRL>
__device__ __forceinline__ int dppi(int v) {
    return __builtin_amdgcn_update_dpp(0, v, CTRL, 0xF, 0xF, true);
}
__device__ __forceinline__ float red16_sum(float v) {
    v += dppf<0xB1>(v); v += dppf<0x4E>(v); v += dppf<0x124>(v); v += dppf<0x128>(v);
    return v;
}
__device__ __forceinline__ float red16_max(float v) {
    v = fmaxf(v, dppf<0xB1>(v)); v = fmaxf(v, dppf<0x4E>(v));
    v = fmaxf(v, dppf<0x124>(v)); v = fmaxf(v, dppf<0x128>(v));
    return v;
}
__device__ __forceinline__ float red32_sum(float v) {
    v = red16_sum(v);
    return v + __shfl_xor(v, 16, 32);
}
__device__ __forceinline__ float red32_max(float v) {
    v = red16_max(v);
    return fmaxf(v, __shfl_xor(v, 16, 32));
}
__device__ __forceinline__ void red32_argmax(float& v, int& id) {
#define AMSTEP(C) { float ov = dppf<C>(v); int oi = dppi<C>(id); \
                    if (ov > v || (ov == v && oi < id)) { v = ov; id = oi; } }
    AMSTEP(0xB1) AMSTEP(0x4E) AMSTEP(0x124) AMSTEP(0x128)
#undef AMSTEP
    float ov = __shfl_xor(v, 16, 32);
    int oi = __shfl_xor(id, 16, 32);
    if (ov > v || (ov == v && oi < id)) { v = ov; id = oi; }
}

// ---------------------------------------------------------------------------
// Generic WMMA GEMM: C[M,N] = act(A[M,KT] * Bt[N,KT]^T + bias) (+resid) (+rmsnorm)
// NORM: requires N == 16*NT; C32 = residual fp32, C16 = rmsnorm(C32)*normw f16.
// VT:   columns >= vtoff are written transposed-only to vtout[b, col-vtoff, S]
//       (V never needs a row-major copy).
// ---------------------------------------------------------------------------
template<int NT, int KT, bool BIAS, bool GELU, bool RESID, bool OUT32, bool OUT16,
         bool NORM, bool VT>
__global__ void gemm_wmma(const f16* __restrict__ A, const f16* __restrict__ Bt,
                          const float* __restrict__ bias, const float* __restrict__ resid,
                          const float* __restrict__ normw,
                          float* __restrict__ C32, f16* __restrict__ C16,
                          f16* __restrict__ vtout, int vtoff,
                          int Mr, int N,
                          long aStride, long bStride, long cStride)
{
    int wid  = threadIdx.x >> 5;
    int lane = threadIdx.x & 31;
    int ln   = lane & 15;
    int g    = lane >> 4;
    int mtiles  = Mr >> 4;
    int cgroups = N / (16 * NT);
    int wglob = blockIdx.x * (blockDim.x >> 5) + wid;
    if (wglob >= mtiles * cgroups) return;
    int mt = wglob / cgroups;
    int cg = wglob % cgroups;
    long batch = blockIdx.y;
    const f16* arow = A + batch * aStride + (size_t)(mt * 16 + ln) * KT;

    v8f acc[NT];
#pragma unroll
    for (int t = 0; t < NT; ++t)
#pragma unroll
        for (int i = 0; i < 8; ++i) acc[t][i] = 0.0f;

#pragma unroll 2
    for (int k0 = 0; k0 < KT; k0 += 32) {
        v8h alo = *(const v8h*)(arow + k0 + g * 8);
        v8h ahi = *(const v8h*)(arow + k0 + 16 + g * 8);
        v8h blo[NT], bhi[NT];
#pragma unroll
        for (int t = 0; t < NT; ++t) {
            const f16* brow = Bt + batch * bStride + (size_t)(cg * NT * 16 + t * 16 + ln) * KT;
            blo[t] = *(const v8h*)(brow + k0 + g * 8);
            bhi[t] = *(const v8h*)(brow + k0 + 16 + g * 8);
        }
        v16h a;
#pragma unroll
        for (int i = 0; i < 8; ++i) { a[i] = alo[i]; a[i + 8] = ahi[i]; }
#pragma unroll
        for (int t = 0; t < NT; ++t) {
            v16h b;
#pragma unroll
            for (int i = 0; i < 8; ++i) { b[i] = blo[t][i]; b[i + 8] = bhi[t][i]; }
            acc[t] = __builtin_amdgcn_wmma_f32_16x16x32_f16(
                false, a, false, b, (short)0, acc[t], false, false);
        }
    }

    // epilogue pass 1: bias/gelu/resid, fp32 out
#pragma unroll
    for (int t = 0; t < NT; ++t) {
        int col = cg * NT * 16 + t * 16 + ln;
#pragma unroll
        for (int r = 0; r < 8; ++r) {
            int row = mt * 16 + r + 8 * g;
            float v = acc[t][r];
            if (BIAS) v += bias[col];
            if (GELU) {
                float xx = v;
                float t3 = xx * xx * xx;
                v = 0.5f * xx * (1.0f + tanhf(0.7978845608028654f * (xx + 0.044715f * t3)));
            }
            size_t idx = (size_t)batch * cStride + (size_t)row * N + col;
            if (RESID) v += resid[idx];
            acc[t][r] = v;
            if (OUT32) C32[idx] = v;
        }
    }
    if (NORM) {
        // per-row rmsnorm over the N=16*NT columns; row (r+8g) lives in one DPP row.
#pragma unroll
        for (int r = 0; r < 8; ++r) {
            float s = 0.0f;
#pragma unroll
            for (int t = 0; t < NT; ++t) s += acc[t][r] * acc[t][r];
            s = red16_sum(s);
            float scn = rsqrtf(s / (float)N + EPS_);
#pragma unroll
            for (int t = 0; t < NT; ++t) {
                int col = t * 16 + ln;
                int row = mt * 16 + r + 8 * g;
                size_t idx = (size_t)row * N + col;
                C16[idx] = (f16)(acc[t][r] * scn * normw[col]);
            }
        }
    } else if (OUT16) {
#pragma unroll
        for (int t = 0; t < NT; ++t) {
            int col = cg * NT * 16 + t * 16 + ln;
#pragma unroll
            for (int r = 0; r < 8; ++r) {
                int row = mt * 16 + r + 8 * g;
                if (VT && col >= vtoff) {
                    int bb = row >> 9;            // row / S_
                    int ss = row & (S_ - 1);
                    vtout[((size_t)bb * D_ + (col - vtoff)) * S_ + ss] = (f16)acc[t][r];
                } else {
                    size_t idx = (size_t)batch * cStride + (size_t)row * N + col;
                    C16[idx] = (f16)acc[t][r];
                }
            }
        }
    }
}

// ---------------------------------------------------------------------------
__global__ void rmsnorm_kernel(const float* __restrict__ x, const float* __restrict__ w,
                               float* __restrict__ o32, f16* __restrict__ o16)
{
    int wid = threadIdx.x >> 5, lane = threadIdx.x & 31;
    int row = blockIdx.x * (blockDim.x >> 5) + wid;
    if (row >= BS_) return;
    const float* xr = x + (size_t)row * D_;
    float a = xr[lane], b = xr[lane + 32];
    float ss = red32_sum(a * a + b * b);
    float sc = rsqrtf(ss / (float)D_ + EPS_);
    float y0 = a * sc * w[lane];
    float y1 = b * sc * w[lane + 32];
    if (o32) { o32[(size_t)row * D_ + lane] = y0; o32[(size_t)row * D_ + lane + 32] = y1; }
    if (o16) { o16[(size_t)row * D_ + lane] = (f16)y0; o16[(size_t)row * D_ + lane + 32] = (f16)y1; }
}

// W[K,N] fp32 -> Wt[N,K] f16
__global__ void transposeW(const float* __restrict__ W, f16* __restrict__ Wt, int K, int N)
{
    int i = blockIdx.x * blockDim.x + threadIdx.x;
    if (i >= K * N) return;
    int k = i / N, n = i % N;
    Wt[(size_t)n * K + k] = (f16)W[i];
}

// ---------------------------------------------------------------------------
// Flash attention, transposed, zero LDS, 32 keys/iteration, mask-split loop.
// ---------------------------------------------------------------------------
template<bool MASK>
__device__ __forceinline__ void attn_step(const f16* __restrict__ kb,
                                          const f16* __restrict__ vb,
                                          int kbase, const v16h& bq, const v8f& cz,
                                          v8f& o, float& mrun, float& lrun,
                                          int qidx, int ln, int g)
{
    v16h ak1, ak2, av;
    {
        v8h lo  = *(const v8h*)(kb + (size_t)(kbase + ln) * QKVS + g * 8);
        v8h l2  = *(const v8h*)(kb + (size_t)(kbase + 16 + ln) * QKVS + g * 8);
        v8h vlo = *(const v8h*)(vb + (size_t)ln * S_ + kbase + g * 8);
        v8h vhi = *(const v8h*)(vb + (size_t)ln * S_ + kbase + 16 + g * 8);
#pragma unroll
        for (int i = 0; i < 8; ++i) {
            ak1[i] = lo[i];  ak1[i + 8] = (f16)0.0f;
            ak2[i] = l2[i];  ak2[i + 8] = (f16)0.0f;
            av[i]  = vlo[i]; av[i + 8]  = vhi[i];
        }
    }
    v8f st1 = __builtin_amdgcn_wmma_f32_16x16x32_f16(false, ak1, false, bq, (short)0, cz, false, false);
    v8f st2 = __builtin_amdgcn_wmma_f32_16x16x32_f16(false, ak2, false, bq, (short)0, cz, false, false);

    float sc1[8], sc2[8];
    float tmax = -3.0e38f;
#pragma unroll
    for (int r = 0; r < 8; ++r) {
        float v1 = st1[r] * 0.25f;              // 1/sqrt(HD)
        float v2 = st2[r] * 0.25f;
        if (MASK) {
            int key1 = kbase + r + 8 * g;
            if (key1 > qidx) v1 = -1.0e30f;
            if (key1 + 16 > qidx) v2 = -1.0e30f;
        }
        sc1[r] = v1; sc2[r] = v2;
        tmax = fmaxf(tmax, fmaxf(v1, v2));
    }
    tmax = fmaxf(tmax, __shfl_xor(tmax, 16, 32));
    float mn = fmaxf(mrun, tmax);
    float corr = __expf(mrun - mn);
    float lsum = 0.0f;
    v16h bp;   // P^T fragment: e<8 -> tile1 (c=8g+e), e>=8 -> tile2
#pragma unroll
    for (int r = 0; r < 8; ++r) {
        float p1 = __expf(sc1[r] - mn);
        float p2 = __expf(sc2[r] - mn);
        lsum += p1 + p2;
        bp[r] = (f16)p1;
        bp[r + 8] = (f16)p2;
    }
    lsum += __shfl_xor(lsum, 16, 32);
    lrun = lrun * corr + lsum;
    mrun = mn;
#pragma unroll
    for (int r = 0; r < 8; ++r) o[r] *= corr;
    o = __builtin_amdgcn_wmma_f32_16x16x32_f16(false, av, false, bp, (short)0, o, false, false);
}

__global__ void attn_flash(const f16* __restrict__ qkv, const f16* __restrict__ vt,
                           f16* __restrict__ out)
{
    int wid = threadIdx.x >> 5, lane = threadIdx.x & 31;
    int ln = lane & 15, g = lane >> 4;
    int b = blockIdx.x / H_, h = blockIdx.x % H_;
    int qt = blockIdx.y + 8 * wid;              // swizzled: balanced causal work
    int qbase = qt * 16;

    const f16* qb = qkv + (size_t)b * S_ * QKVS + h * HD_;          // Q cols 0..63
    const f16* kb = qkv + (size_t)b * S_ * QKVS + D_ + h * HD_;     // K cols 64..127
    const f16* vb = vt + ((size_t)b * D_ + h * HD_) * S_;

    v16h bq;   // B operand: Q^T (col = query qbase+ln, c = hd padded)
    {
        v8h lo = *(const v8h*)(qb + (size_t)(qbase + ln) * QKVS + g * 8);
#pragma unroll
        for (int i = 0; i < 8; ++i) { bq[i] = lo[i]; bq[i + 8] = (f16)0.0f; }
    }
    v8f o, cz;
#pragma unroll
    for (int r = 0; r < 8; ++r) { o[r] = 0.0f; cz[r] = 0.0f; }
    float mrun = -3.0e38f, lrun = 0.0f;
    int qidx = qbase + ln;

    int nfull = qbase >> 5;                     // fully-unmasked 32-key blocks
    for (int kp = 0; kp < nfull; ++kp)
        attn_step<false>(kb, vb, kp * 32, bq, cz, o, mrun, lrun, qidx, ln, g);
    attn_step<true>(kb, vb, nfull * 32, bq, cz, o, mrun, lrun, qidx, ln, g);

    float inv = 1.0f / lrun;
    v8h res;
#pragma unroll
    for (int r = 0; r < 8; ++r) res[r] = (f16)(o[r] * inv);
    *(v8h*)(out + ((size_t)b * S_ + qbase + ln) * D_ + h * HD_ + g * 8) = res;
}

// normalize mem_addr rows: [M,A] -> [M,A]
__global__ void addr_norm(const float* __restrict__ ma, float* __restrict__ an)
{
    int wid = threadIdx.x >> 5, lane = threadIdx.x & 31;
    int m = blockIdx.x * (blockDim.x >> 5) + wid;
    if (m >= M_) return;
    float v = ma[(size_t)m * A_ + lane];
    float ss = red32_sum(v * v);
    an[(size_t)m * A_ + lane] = v / fmaxf(sqrtf(ss), 1e-12f);
}

// ---------------------------------------------------------------------------
// Top-K memory read: one wave per (b,s). lane owns m=lane, m=lane+32.
// ---------------------------------------------------------------------------
__global__ void memread_topk(const float* __restrict__ rq, const float* __restrict__ addrn,
                             const float* __restrict__ mv, f16* __restrict__ rv16)
{
    __shared__ __align__(16) float qn[4][32];
    int wid = threadIdx.x >> 5, lane = threadIdx.x & 31;
    int row = blockIdx.x * 4 + wid;
    if (row >= BS_) return;
    int b = row / S_;

    float qa = rq[(size_t)row * A_ + lane];
    float ss = red32_sum(qa * qa);
    qn[wid][lane] = qa / fmaxf(sqrtf(ss), 1e-12f);
    __asm volatile("s_wait_dscnt 0x0" ::: "memory");

    float sc0 = 0.0f, sc1 = 0.0f;
    const float4* a0 = (const float4*)(addrn + (size_t)lane * A_);
    const float4* a1 = (const float4*)(addrn + (size_t)(lane + 32) * A_);
    const float4* qv = (const float4*)qn[wid];
#pragma unroll
    for (int a4 = 0; a4 < A_ / 4; ++a4) {
        float4 q4 = qv[a4], x0 = a0[a4], x1 = a1[a4];
        sc0 += q4.x * x0.x + q4.y * x0.y + q4.z * x0.z + q4.w * x0.w;
        sc1 += q4.x * x1.x + q4.y * x1.y + q4.z * x1.z + q4.w * x1.w;
    }
    sc0 *= 4.0f; sc1 *= 4.0f;                       // 1/RT

    float tv[KTOP]; int ti[KTOP];
    for (int kk = 0; kk < KTOP; ++kk) {
        float v; int id;
        if (sc0 >= sc1) { v = sc0; id = lane; } else { v = sc1; id = lane + 32; }
        red32_argmax(v, id);
        tv[kk] = v; ti[kk] = id;
        if (id == lane) sc0 = -3.0e38f;
        if (id == lane + 32) sc1 = -3.0e38f;
    }
    float mx = tv[0], sum = 0.0f, w[KTOP];
    for (int kk = 0; kk < KTOP; ++kk) { w[kk] = __expf(tv[kk] - mx); sum += w[kk]; }
    float isum = 1.0f / sum;

    float r0 = 0.0f, r1 = 0.0f;
    const float* mvb = mv + (size_t)b * M_ * D_;
    for (int kk = 0; kk < KTOP; ++kk) {
        const float* mr = mvb + (size_t)ti[kk] * D_;
        float wk = w[kk] * isum;
        r0 += wk * mr[lane];
        r1 += wk * mr[lane + 32];
    }
    rv16[(size_t)row * D_ + lane] = (f16)r0;
    rv16[(size_t)row * D_ + lane + 32] = (f16)r1;
}

// ---------------------------------------------------------------------------
// Memory-write scoring: full softmax over M=64, sigmoid gate, write w^T [b,M,S] f16
// ---------------------------------------------------------------------------
__global__ void memwrite_scores(const float* __restrict__ wqo, const float* __restrict__ addrn,
                                const f16* __restrict__ xw16, const float* __restrict__ gw,
                                const float* __restrict__ gb, f16* __restrict__ wt16)
{
    __shared__ __align__(16) float qn[4][32];
    int wid = threadIdx.x >> 5, lane = threadIdx.x & 31;
    int row = blockIdx.x * 4 + wid;
    if (row >= BS_) return;
    int b = row / S_, s = row % S_;

    float qa = wqo[(size_t)row * A_ + lane];
    float ss = red32_sum(qa * qa);
    qn[wid][lane] = qa / fmaxf(sqrtf(ss), 1e-12f);
    __asm volatile("s_wait_dscnt 0x0" ::: "memory");

    float sc0 = 0.0f, sc1 = 0.0f;
    const float4* a0 = (const float4*)(addrn + (size_t)lane * A_);
    const float4* a1 = (const float4*)(addrn + (size_t)(lane + 32) * A_);
    const float4* qv = (const float4*)qn[wid];
#pragma unroll
    for (int a4 = 0; a4 < A_ / 4; ++a4) {
        float4 q4 = qv[a4], x0 = a0[a4], x1 = a1[a4];
        sc0 += q4.x * x0.x + q4.y * x0.y + q4.z * x0.z + q4.w * x0.w;
        sc1 += q4.x * x1.x + q4.y * x1.y + q4.z * x1.z + q4.w * x1.w;
    }
    sc0 *= 4.0f; sc1 *= 4.0f;                       // 1/WT

    float mx = red32_max(fmaxf(sc0, sc1));
    float e0 = __expf(sc0 - mx), e1 = __expf(sc1 - mx);
    float sum = red32_sum(e0 + e1);

    float xa = (float)xw16[(size_t)row * D_ + lane];
    float xb = (float)xw16[(size_t)row * D_ + lane + 32];
    float gd = red32_sum(xa * gw[lane] + xb * gw[lane + 32]);
    float gate = 1.0f / (1.0f + __expf(-(gd + gb[0])));

    float w0 = e0 / sum * gate;
    float w1 = e1 / sum * gate;
    wt16[((size_t)b * M_ + lane) * S_ + s] = (f16)w0;
    wt16[((size_t)b * M_ + lane + 32) * S_ + s] = (f16)w1;
}

// suw[b,m] = sum_s wt16[b,m,s]; one wave per (b,m), b128 loads
__global__ void rowsum_wt(const f16* __restrict__ wt16, float* __restrict__ suw)
{
    int wid = threadIdx.x >> 5, lane = threadIdx.x & 31;
    int row = blockIdx.x * (blockDim.x >> 5) + wid;
    if (row >= B_ * M_) return;
    const f16* r = wt16 + (size_t)row * S_;
    v8h a = *(const v8h*)(r + lane * 16);
    v8h b = *(const v8h*)(r + lane * 16 + 8);
    float s = 0.0f;
#pragma unroll
    for (int i = 0; i < 8; ++i) s += (float)a[i] + (float)b[i];
    s = red32_sum(s);
    if (lane == 0) suw[row] = s;
}

// new_mem = mem*(1-sg) + (sus/clip(suw,1e-6))*sg
__global__ void mem_combine(const float* __restrict__ mv, const float* __restrict__ sus,
                            const float* __restrict__ suw, float* __restrict__ out)
{
    int i = blockIdx.x * blockDim.x + threadIdx.x;
    if (i >= B_ * M_ * D_) return;
    int bm = i / D_;
    float sw = suw[bm];
    float sg = 1.0f - __expf(-sw);
    float upd = sus[i] / fmaxf(sw, 1e-6f);
    out[i] = mv[i] * (1.0f - sg) + upd * sg;
}

// ---------------------------------------------------------------------------
extern "C" void kernel_launch(void* const* d_in, const int* in_sizes, int n_in,
                              void* d_out, int out_size, void* d_ws, size_t ws_size,
                              hipStream_t stream)
{
    const float* in_x    = (const float*)d_in[0];
    const float* in_mv   = (const float*)d_in[1];
    const float* attn_nw = (const float*)d_in[2];
    const float* wq_in   = (const float*)d_in[3];
    const float* wk_in   = (const float*)d_in[4];
    const float* wv_in   = (const float*)d_in[5];
    const float* wo_in   = (const float*)d_in[6];
    const float* read_nw = (const float*)d_in[7];
    const float* rq_in   = (const float*)d_in[8];
    const float* ro_in   = (const float*)d_in[9];
    const float* ffn_nw  = (const float*)d_in[10];
    const float* w1_in   = (const float*)d_in[11];
    const float* b1_in   = (const float*)d_in[12];
    const float* w2_in   = (const float*)d_in[13];
    const float* b2_in   = (const float*)d_in[14];
    const float* out_nw  = (const float*)d_in[15];
    const float* wrt_nw  = (const float*)d_in[16];
    const float* mem_ad  = (const float*)d_in[17];
    const float* wqp_in  = (const float*)d_in[18];
    const float* wvp_in  = (const float*)d_in[19];
    const float* gate_w  = (const float*)d_in[20];
    const float* gate_b  = (const float*)d_in[21];

    char* p = (char*)d_ws;
    auto alloc = [&](size_t bytes) -> char* {
        char* r = p;
        p += (bytes + 255) & ~(size_t)255;
        return r;
    };
    float* x32   = (float*)alloc((size_t)BS_ * D_ * 4);
    f16* xn16    = (f16*)alloc((size_t)BS_ * D_ * 2);
    f16* qkv16   = (f16*)alloc((size_t)BS_ * QKVS * 2);
    f16* vt16    = (f16*)alloc((size_t)BS_ * D_ * 2);
    f16* t16     = (f16*)alloc((size_t)BS_ * D_ * 2);
    f16* ffnh16  = (f16*)alloc((size_t)BS_ * HID_ * 2);
    float* f32a  = (float*)alloc((size_t)BS_ * A_ * 4);
    float* addrn = (float*)alloc((size_t)M_ * A_ * 4);
    f16* wtb16   = (f16*)alloc((size_t)B_ * M_ * S_ * 2);
    float* sus32 = (float*)alloc((size_t)B_ * M_ * D_ * 4);
    float* suw32 = (float*)alloc((size_t)B_ * M_ * 4);

    f16 *wqkvT[NB_], *woT[NB_], *rqT[NB_], *roT[NB_], *w1T[NB_], *w2T[NB_];
    for (int i = 0; i < NB_; ++i) {
        wqkvT[i] = (f16*)alloc(QKVS * D_ * 2);      // [192 rows][64 K]
        woT[i]   = (f16*)alloc(D_ * D_ * 2);
        rqT[i]   = (f16*)alloc(D_ * A_ * 2);
        roT[i]   = (f16*)alloc(D_ * D_ * 2);
        w1T[i]   = (f16*)alloc(D_ * HID_ * 2);
        w2T[i]   = (f16*)alloc(HID_ * D_ * 2);
    }
    f16* wqpT = (f16*)alloc(D_ * A_ * 2);
    f16* wvpT = (f16*)alloc(D_ * D_ * 2);

    hipMemcpyAsync(x32, in_x, (size_t)BS_ * D_ * 4, hipMemcpyDeviceToDevice, stream);

    dim3 blk(128);
    addr_norm<<<dim3(M_ / 4), blk, 0, stream>>>(mem_ad, addrn);

    int nDD = D_ * D_, nDA = D_ * A_, nDH = D_ * HID_;
    for (int i = 0; i < NB_; ++i) {
        transposeW<<<(nDD + 255) / 256, 256, 0, stream>>>(wq_in + (size_t)i * nDD, wqkvT[i], D_, D_);
        transposeW<<<(nDD + 255) / 256, 256, 0, stream>>>(wk_in + (size_t)i * nDD, wqkvT[i] + nDD, D_, D_);
        transposeW<<<(nDD + 255) / 256, 256, 0, stream>>>(wv_in + (size_t)i * nDD, wqkvT[i] + 2 * nDD, D_, D_);
        transposeW<<<(nDD + 255) / 256, 256, 0, stream>>>(wo_in + (size_t)i * nDD, woT[i], D_, D_);
        transposeW<<<(nDA + 255) / 256, 256, 0, stream>>>(rq_in + (size_t)i * nDA, rqT[i], D_, A_);
        transposeW<<<(nDD + 255) / 256, 256, 0, stream>>>(ro_in + (size_t)i * nDD, roT[i], D_, D_);
        transposeW<<<(nDH + 255) / 256, 256, 0, stream>>>(w1_in + (size_t)i * nDH, w1T[i], D_, HID_);
        transposeW<<<(nDH + 255) / 256, 256, 0, stream>>>(w2_in + (size_t)i * nDH, w2T[i], HID_, D_);
    }
    transposeW<<<(nDA + 255) / 256, 256, 0, stream>>>(wqp_in, wqpT, D_, A_);
    transposeW<<<(nDD + 255) / 256, 256, 0, stream>>>(wvp_in, wvpT, D_, D_);

    int gQKV = (BS_ / 16 * (QKVS / 64) + 3) / 4;        // N=192, NT=4 -> 1536 blocks
    int gDD  = (BS_ / 16 * (D_ / 64) + 3) / 4;          // N=64,  NT=4 -> 512 blocks
    int gRQ  = (BS_ / 16 * (A_ / 32) + 3) / 4;          // N=32,  NT=2 -> 512 blocks
    int gF1  = (BS_ / 16 * (HID_ / 64) + 3) / 4;        // N=256, NT=4 -> 2048 blocks

    // initial norm for block 0
    rmsnorm_kernel<<<BS_ / 4, blk, 0, stream>>>(x32, attn_nw, nullptr, xn16);

    for (int i = 0; i < NB_; ++i) {
        // ---- attention: QKV fused GEMM, V written transposed in-epilogue ----
        gemm_wmma<4, 64, false, false, false, false, true, false, true><<<gQKV, blk, 0, stream>>>(
            xn16, wqkvT[i], nullptr, nullptr, nullptr, nullptr, qkv16, vt16, 2 * D_,
            BS_, QKVS, 0, 0, 0);
        attn_flash<<<dim3(B_ * H_, S_ / 64), blk, 0, stream>>>(qkv16, vt16, t16);
        gemm_wmma<4, 64, false, false, true, true, false, true, false><<<gDD, blk, 0, stream>>>(
            t16, woT[i], nullptr, x32, read_nw + (size_t)i * D_, x32, xn16, nullptr, 0,
            BS_, D_, 0, 0, 0);
        // ---- memory read ----
        gemm_wmma<2, 64, false, false, false, true, false, false, false><<<gRQ, blk, 0, stream>>>(
            xn16, rqT[i], nullptr, nullptr, nullptr, f32a, nullptr, nullptr, 0,
            BS_, A_, 0, 0, 0);
        memread_topk<<<BS_ / 4, blk, 0, stream>>>(f32a, addrn, in_mv, t16);
        gemm_wmma<4, 64, false, false, true, true, false, true, false><<<gDD, blk, 0, stream>>>(
            t16, roT[i], nullptr, x32, ffn_nw + (size_t)i * D_, x32, xn16, nullptr, 0,
            BS_, D_, 0, 0, 0);
        // ---- FFN ----
        gemm_wmma<4, 64, true, true, false, false, true, false, false><<<gF1, blk, 0, stream>>>(
            xn16, w1T[i], b1_in + (size_t)i * HID_, nullptr, nullptr, nullptr, ffnh16, nullptr, 0,
            BS_, HID_, 0, 0, 0);
        if (i < NB_ - 1) {
            gemm_wmma<4, 256, true, false, true, true, false, true, false><<<gDD, blk, 0, stream>>>(
                ffnh16, w2T[i], b2_in + (size_t)i * D_, x32, attn_nw + (size_t)(i + 1) * D_,
                x32, xn16, nullptr, 0, BS_, D_, 0, 0, 0);
        } else {
            gemm_wmma<4, 256, true, false, true, true, false, false, false><<<gDD, blk, 0, stream>>>(
                ffnh16, w2T[i], b2_in + (size_t)i * D_, x32, nullptr, x32, nullptr, nullptr, 0,
                BS_, D_, 0, 0, 0);
        }
    }

    float* xout = (float*)d_out;
    rmsnorm_kernel<<<BS_ / 4, blk, 0, stream>>>(x32, out_nw, xout, nullptr);
    rmsnorm_kernel<<<BS_ / 4, blk, 0, stream>>>(xout, wrt_nw, nullptr, xn16);
    gemm_wmma<2, 64, false, false, false, true, false, false, false><<<gRQ, blk, 0, stream>>>(
        xn16, wqpT, nullptr, nullptr, nullptr, f32a, nullptr, nullptr, 0,
        BS_, A_, 0, 0, 0);
    // write_v projection: output written directly transposed (vtoff = 0)
    gemm_wmma<4, 64, false, false, false, false, true, false, true><<<gDD, blk, 0, stream>>>(
        xn16, wvpT, nullptr, nullptr, nullptr, nullptr, nullptr, vt16, 0,
        BS_, D_, 0, 0, 0);
    memwrite_scores<<<BS_ / 4, blk, 0, stream>>>(f32a, addrn, xn16, gate_w, gate_b, wtb16);
    rowsum_wt<<<(B_ * M_) / 4, blk, 0, stream>>>(wtb16, suw32);
    // batched: sus[b] = wt[b][M,S] @ vu[b][S,D]  (Bt = vut[b][D,S])
    gemm_wmma<4, 512, false, false, false, true, false, false, false><<<dim3(1, B_), blk, 0, stream>>>(
        wtb16, vt16, nullptr, nullptr, nullptr, sus32, nullptr, nullptr, 0,
        M_, D_, (long)M_ * S_, (long)D_ * S_, (long)M_ * D_);
    mem_combine<<<(B_ * M_ * D_ + 255) / 256, 256, 0, stream>>>(
        in_mv, sus32, suw32, xout + (size_t)BS_ * D_);
}
